// LocalAttention_50319836840434
// MI455X (gfx1250) — matrix operand order repaired and checked by reading, not compiled, not run on hardware
//
#include <hip/hip_runtime.h>
#include <stddef.h>

// ---------------------------------------------------------------------------
// LocalAttention for MI455X (gfx1250): bf16 WMMA GEMMs + windowed attention.
// Shapes: B=2, N=2048, E=1024, H=16, d=64, window=128.
// ---------------------------------------------------------------------------

typedef __bf16 bf16;
typedef __attribute__((ext_vector_type(16))) __bf16 v16bf;
typedef __attribute__((ext_vector_type(8)))  __bf16 v8bf;
typedef __attribute__((ext_vector_type(4)))  __bf16 v4bf;
typedef __attribute__((ext_vector_type(8)))  float  v8f;

#define SEQ   2048
#define EMBD  1024
#define NHEAD 16
#define HDIM  64
#define WIN   128
#define NBAT  2
#define MROWS (NBAT * SEQ)       // 4096 GEMM rows
#define SSTR  292                // LDS row stride for the S strip (bank-friendly)
#define NKT   18                 // key tiles of 16 covering [q0-128, q0+15+128] (+pad)

static __device__ inline v8f wmma_bf16(v16bf a, v16bf b, v8f c) {
  return __builtin_amdgcn_wmma_f32_16x16x32_bf16(false, a, false, b, (short)0, c,
                                                 false, false);
}

// ---------------- elementwise converters ----------------------------------

__global__ void k_cvt_bf16(const float4* __restrict__ in, v4bf* __restrict__ out,
                           int n4) {
  int i = blockIdx.x * blockDim.x + threadIdx.x;
  if (i < n4) {
    const float4 f = in[i];
    v4bf o;
    o[0] = (bf16)f.x; o[1] = (bf16)f.y; o[2] = (bf16)f.z; o[3] = (bf16)f.w;
    out[i] = o;
  }
}

// W f32 [K][N] row-major  ->  Wt bf16 [N][K]  (so B-fragments read contiguous K)
__global__ void k_cvt_transpose(const float* __restrict__ in,
                                bf16* __restrict__ out) {
  int i = blockIdx.x * blockDim.x + threadIdx.x;  // over EMBD*EMBD
  if (i < EMBD * EMBD) {
    int n = i >> 10;        // output row   (original column)
    int k = i & 1023;       // output col   (original row)
    out[i] = (bf16)in[k * EMBD + n];
  }
}

// ---------------- bf16 WMMA GEMM: C[4096x1024] = A @ Wt^T + bias ------------
// FLAG 0: store bf16 row-major [m][n]                 (Q, K)
// FLAG 1: store bf16 head-transposed Vt[b][h][d][q]   (V)
// FLAG 2: store f32 row-major [m][n]                  (final output)
// Wave tile 64x64 (16 WMMAs / K-step); block = 4 waves (2x2) -> 128x128 tile.
template <int FLAG>
__global__ __launch_bounds__(128) void k_gemm_bf16(
    const bf16* __restrict__ A, const bf16* __restrict__ Bt,
    const float* __restrict__ bias, void* __restrict__ outp) {
  const int lane = threadIdx.x & 31;
  const int wv   = threadIdx.x >> 5;
  const int hh   = lane >> 4;     // half-wave index
  const int l16  = lane & 15;
  const int mBase = blockIdx.y * 128 + (wv >> 1) * 64;
  const int nBase = blockIdx.x * 128 + (wv & 1) * 64;

  v8f acc[4][4];
#pragma unroll
  for (int mi = 0; mi < 4; ++mi)
#pragma unroll
    for (int ni = 0; ni < 4; ++ni)
#pragma unroll
      for (int j = 0; j < 8; ++j) acc[mi][ni][j] = 0.0f;

  for (int k = 0; k < EMBD; k += 32) {
    v16bf afr[4];
#pragma unroll
    for (int mi = 0; mi < 4; ++mi) {
      const bf16* ap = A + (size_t)(mBase + mi * 16 + l16) * EMBD + k;
      afr[mi] = *(const v16bf*)(ap + hh * 16);            // K = k+hh*16 .. +15
    }
    v16bf bfr[4];
#pragma unroll
    for (int ni = 0; ni < 4; ++ni) {
      bfr[ni] = *(const v16bf*)(Bt + (size_t)(nBase + ni * 16 + l16) * EMBD + k +
                                hh * 16);                  // K = k+hh*16 .. +15
    }
#pragma unroll
    for (int mi = 0; mi < 4; ++mi)
#pragma unroll
      for (int ni = 0; ni < 4; ++ni)
        acc[mi][ni] = wmma_bf16(afr[mi], bfr[ni], acc[mi][ni]);
  }

#pragma unroll
  for (int mi = 0; mi < 4; ++mi)
#pragma unroll
    for (int ni = 0; ni < 4; ++ni) {
      const int col = nBase + ni * 16 + l16;
      const float bs = bias[col];
#pragma unroll
      for (int j = 0; j < 8; ++j) {
        const int m = mBase + mi * 16 + j + 8 * hh;
        const float v = acc[mi][ni][j] + bs;
        if (FLAG == 0) {
          ((bf16*)outp)[(size_t)m * EMBD + col] = (bf16)v;
        } else if (FLAG == 2) {
          ((float*)outp)[(size_t)m * EMBD + col] = v;
        } else {  // Vt[b][h][d][q]
          const int b = m >> 11, q = m & (SEQ - 1);
          const int h = col >> 6, dd = col & 63;
          ((bf16*)outp)[(((size_t)(b * NHEAD + h) * HDIM + dd) * SEQ) + q] =
              (bf16)v;
        }
      }
    }
}

// ---------------- windowed attention, one wave per 16-query tile -----------
// Q, K: bf16 [b][q][h*64+d] ; Vt: bf16 [b][h][d][q] ; O: bf16 [b][q][h*64+d]
__global__ __launch_bounds__(32) void k_attn(const bf16* __restrict__ Q,
                                             const bf16* __restrict__ Kc,
                                             const bf16* __restrict__ Vt,
                                             bf16* __restrict__ O) {
  __shared__ float S[16 * SSTR];
  __shared__ float rsumInv[16];

  const int lane = threadIdx.x;
  const int hh = lane >> 4, l16 = lane & 15;
  const int bid = blockIdx.x;
  const int qt = bid & 127;
  const int h  = (bid >> 7) & (NHEAD - 1);
  const int b  = bid >> 11;
  const int kstart = qt * 16 - WIN;

  // --- Q fragments for the two K-steps over d (0..31, 32..63) ---
  const bf16* qp = Q + (size_t)(b * SEQ + qt * 16 + l16) * EMBD + h * HDIM;
  v16bf qa[2];
#pragma unroll
  for (int s = 0; s < 2; ++s) {
    qa[s] = *(const v16bf*)(qp + 32 * s + hh * 16);
  }

  // --- pass 1: S = (Q K^T) / 32, masked, into LDS strip ---
  for (int kt = 0; kt < NKT; ++kt) {
    const int kg = kstart + kt * 16 + l16;                 // this lane's key
    const int kkc = kg < 0 ? 0 : (kg >= SEQ ? SEQ - 1 : kg);
    const bf16* kp = Kc + (size_t)(b * SEQ + kkc) * EMBD + h * HDIM;
    const v16bf kb0 = *(const v16bf*)(kp + hh * 16);       // d = hh*16 .. +15
    const v16bf kb1 = *(const v16bf*)(kp + 32 + hh * 16);  // d = 32+hh*16 ..
    v8f c;
#pragma unroll
    for (int j = 0; j < 8; ++j) c[j] = 0.0f;
    c = wmma_bf16(qa[0], kb0, c);
    c = wmma_bf16(qa[1], kb1, c);
#pragma unroll
    for (int j = 0; j < 8; ++j) {
      const int qg = qt * 16 + j + 8 * hh;
      int dif = qg - kg; dif = dif < 0 ? -dif : dif;
      const bool bad = (kg < 0) | (kg >= SEQ) | (dif > WIN);
      const float v = bad ? -3.0e37f : c[j] * 0.03125f;    // /= sqrt(1024)
      S[(j + 8 * hh) * SSTR + kt * 16 + l16] = v;
    }
  }
  __syncthreads();

  // --- pass 2: row softmax (each half-wave scans half the 288 columns) ---
  {
    const int row = l16;
    const int c0 = hh * 144, c1 = c0 + 144;
    float m = -3.4e38f;
    for (int c = c0; c < c1; ++c) m = fmaxf(m, S[row * SSTR + c]);
    m = fmaxf(m, __shfl_xor(m, 16, 32));
    float sum = 0.0f;
    for (int c = c0; c < c1; ++c) {
      const float e = __expf(S[row * SSTR + c] - m);
      S[row * SSTR + c] = e;
      sum += e;
    }
    sum += __shfl_xor(sum, 16, 32);
    if (hh == 0) rsumInv[row] = 1.0f / sum;
  }
  __syncthreads();

  // --- pass 3: O = P V  (9 chunks of 32 keys, 4 d-column tiles) ---
  v8f acc[4];
#pragma unroll
  for (int t = 0; t < 4; ++t)
#pragma unroll
    for (int j = 0; j < 8; ++j) acc[t][j] = 0.0f;

  for (int c = 0; c < 9; ++c) {
    v16bf pa;
    const float* sp = &S[l16 * SSTR + c * 32];
#pragma unroll
    for (int g = 0; g < 8; ++g) {
      pa[g]     = (bf16)sp[hh * 16 + g];           // keys c*32 + hh*16 .. +7
      pa[8 + g] = (bf16)sp[hh * 16 + 8 + g];       // keys c*32+hh*16+8 .. +15
    }
    const int kk = kstart + c * 32 + hh * 16;      // always multiple of 16
    const int kkb = (kk < 0 || kk + 16 > SEQ) ? 0 : kk;  // P==0 when clamped
#pragma unroll
    for (int t = 0; t < 4; ++t) {
      const bf16* vp =
          Vt + ((size_t)(b * NHEAD + h) * HDIM + t * 16 + l16) * SEQ + kkb;
      const v16bf vb = *(const v16bf*)vp;
      acc[t] = wmma_bf16(pa, vb, acc[t]);
    }
  }

#pragma unroll
  for (int t = 0; t < 4; ++t)
#pragma unroll
    for (int j = 0; j < 8; ++j) {
      const int qg = qt * 16 + j + 8 * hh;
      const float o = acc[t][j] * rsumInv[j + 8 * hh];
      O[(size_t)(b * SEQ + qg) * EMBD + h * HDIM + t * 16 + l16] = (bf16)o;
    }
}

// ---------------------------------------------------------------------------

extern "C" void kernel_launch(void* const* d_in, const int* in_sizes, int n_in,
                              void* d_out, int out_size, void* d_ws,
                              size_t ws_size, hipStream_t stream) {
  const float* x  = (const float*)d_in[0];
  const float* wq = (const float*)d_in[1];
  const float* bq = (const float*)d_in[2];
  const float* wk = (const float*)d_in[3];
  const float* bk = (const float*)d_in[4];
  const float* wv = (const float*)d_in[5];
  const float* bv = (const float*)d_in[6];
  const float* wo = (const float*)d_in[7];
  const float* bo = (const float*)d_in[8];

  char* ws = (char*)d_ws;
  const size_t MB = 1u << 20;
  bf16* xb  = (bf16*)(ws);              //  8 MB  bf16 x [b][q][e]
  bf16* wqt = (bf16*)(ws + 8 * MB);     //  2 MB  each, transposed bf16 weights
  bf16* wkt = (bf16*)(ws + 10 * MB);
  bf16* wvt = (bf16*)(ws + 12 * MB);
  bf16* wot = (bf16*)(ws + 14 * MB);
  bf16* Qb  = (bf16*)(ws + 16 * MB);    //  8 MB
  bf16* Kb  = (bf16*)(ws + 24 * MB);    //  8 MB
  bf16* Vt  = (bf16*)(ws + 32 * MB);    //  8 MB  [b][h][d][q]
  bf16* Ob  = xb;                       //  reuse: x bf16 is dead after QKV

  const int nX4 = (MROWS * EMBD) / 4;
  k_cvt_bf16<<<(nX4 + 255) / 256, 256, 0, stream>>>((const float4*)x,
                                                    (v4bf*)xb, nX4);
  const int nW = EMBD * EMBD;
  k_cvt_transpose<<<(nW + 255) / 256, 256, 0, stream>>>(wq, wqt);
  k_cvt_transpose<<<(nW + 255) / 256, 256, 0, stream>>>(wk, wkt);
  k_cvt_transpose<<<(nW + 255) / 256, 256, 0, stream>>>(wv, wvt);
  k_cvt_transpose<<<(nW + 255) / 256, 256, 0, stream>>>(wo, wot);

  dim3 ggrid(EMBD / 128, MROWS / 128);
  k_gemm_bf16<0><<<ggrid, 128, 0, stream>>>(xb, wqt, bq, (void*)Qb);
  k_gemm_bf16<0><<<ggrid, 128, 0, stream>>>(xb, wkt, bk, (void*)Kb);
  k_gemm_bf16<1><<<ggrid, 128, 0, stream>>>(xb, wvt, bv, (void*)Vt);

  k_attn<<<NBAT * NHEAD * (SEQ / 16), 32, 0, stream>>>(Qb, Kb, Vt, Ob);

  k_gemm_bf16<2><<<ggrid, 128, 0, stream>>>(Ob, wot, bo, d_out);
}
